// WeightedEnsembleAggregator_9603546874234
// MI455X (gfx1250) — matrix operand order repaired and checked by reading, not compiled, not run on hardware
//
#include <hip/hip_runtime.h>
#include <hip/hip_bf16.h>
#include <math.h>

typedef __attribute__((ext_vector_type(16))) __bf16 v16bf;
typedef __attribute__((ext_vector_type(8)))  float  v8f;

#define INPUT_DIM  256
#define HIDDEN     1024
#define NUM_GROUPS 1000
#define FEAT_STRIDE 512
#define OUT_STRIDE  1280

union Frag {
    v16bf v;
    uint4 q[2];
};

// ---------------- async global -> LDS copy (CDNA5) ----------------
// GLOBAL_LOAD_ASYNC_TO_LDS_B128: per-lane copy of 16B from global to LDS,
// tracked with ASYNCcnt.
__device__ __forceinline__ void async_copy_b128(void* lds_ptr, const void* gaddr) {
    unsigned lds_off = (unsigned)(size_t)lds_ptr;   // low 32 bits = LDS offset
    asm volatile("global_load_async_to_lds_b128 %0, %1, off"
                 :: "v"(lds_off), "v"(gaddr)
                 : "memory");
}
__device__ __forceinline__ void async_wait_all() {
    asm volatile("s_wait_asynccnt 0x0" ::: "memory");
}

// ---------------- ordered-float <-> uint key for atomic max ----------------
__device__ __forceinline__ unsigned fkey(float f) {
    unsigned b = __float_as_uint(f);
    return (b & 0x80000000u) ? ~b : (b | 0x80000000u);
}
__device__ __forceinline__ float fdecode(unsigned k) {
    unsigned b = (k & 0x80000000u) ? (k & 0x7FFFFFFFu) : ~k;
    return __uint_as_float(b);
}

// ---------------- 0) weights: transpose + f32 -> bf16 ----------------
__global__ void prep_weights(const float* __restrict__ W1,
                             const float* __restrict__ W2,
                             const float* __restrict__ W3,
                             __bf16* __restrict__ W1t,
                             __bf16* __restrict__ W2t,
                             __bf16* __restrict__ W3t) {
    int idx = blockIdx.x * blockDim.x + threadIdx.x;
    const int SZ12 = INPUT_DIM * HIDDEN;      // 262144
    if (idx < SZ12) {
        int k = idx & (INPUT_DIM - 1);
        int n = idx >> 8;
        W1t[idx] = (__bf16)W1[(size_t)k * HIDDEN + n];
    } else if (idx < 2 * SZ12) {
        int o = idx - SZ12;
        int k = o & (INPUT_DIM - 1);
        int n = o >> 8;
        W2t[o] = (__bf16)W2[(size_t)k * HIDDEN + n];
    } else if (idx < 2 * SZ12 + HIDDEN * HIDDEN) {
        int o = idx - 2 * SZ12;
        int k = o & (HIDDEN - 1);
        int n = o >> 10;
        W3t[o] = (__bf16)W3[(size_t)k * HIDDEN + n];
    }
}

// ---------------- 1) LayerNorm -> bf16 (one wave per row) ----------------
__global__ void ln_kernel(const float* __restrict__ feat,
                          const float* __restrict__ g,
                          const float* __restrict__ b,
                          __bf16* __restrict__ xn, int N) {
    int row  = blockIdx.x * 8 + (threadIdx.x >> 5);
    int lane = threadIdx.x & 31;
    if (row >= N) return;
    const float* x = feat + (size_t)row * FEAT_STRIDE;
    int base = lane * 8;
    float v[8];
    float s = 0.f;
#pragma unroll
    for (int i = 0; i < 8; ++i) { v[i] = x[base + i]; s += v[i]; }
    for (int off = 16; off; off >>= 1) s += __shfl_xor(s, off, 32);
    float mu = s * (1.0f / INPUT_DIM);
    float vs = 0.f;
#pragma unroll
    for (int i = 0; i < 8; ++i) { float d = v[i] - mu; vs += d * d; }
    for (int off = 16; off; off >>= 1) vs += __shfl_xor(vs, off, 32);
    float rsig = rsqrtf(vs * (1.0f / INPUT_DIM) + 1e-5f);
    __bf16* o = xn + (size_t)row * INPUT_DIM;
#pragma unroll
    for (int i = 0; i < 8; ++i) {
        float y = (v[i] - mu) * rsig * g[base + i] + b[base + i];
        o[base + i] = (__bf16)y;
    }
}

// ---------------- 2) GEMM1 (dual) + SiLU gate -> U (bf16) ----------------
// grid = (N/16, 4), block = 256 (8 waves). Each wave computes TWO adjacent
// 16x16 N-tiles for BOTH W1 and W2 (4 wmma per A-fragment load), then stores
// u = a * sigmoid(b) as bf16.
__global__ void gemm1_silu(const __bf16* __restrict__ xn,
                           const __bf16* __restrict__ W1t,
                           const __bf16* __restrict__ W2t,
                           const float* __restrict__ b1,
                           const float* __restrict__ b2,
                           __bf16* __restrict__ U) {
    __shared__ uint4 At[512];                       // 16 x 256 bf16 = 8 KB
    const int tid = threadIdx.x, lane = tid & 31, wave = tid >> 5;
    const int m0 = blockIdx.x * 16;

    const uint4* src = reinterpret_cast<const uint4*>(xn + (size_t)m0 * INPUT_DIM);
    async_copy_b128(&At[tid],       src + tid);
    async_copy_b128(&At[tid + 256], src + tid + 256);
    async_wait_all();
    __syncthreads();

    const __bf16* A = reinterpret_cast<const __bf16*>(At);
    const int n0   = (blockIdx.y * 8 + wave) * 32;  // 32 columns per wave
    const int nc0  = n0 + (lane & 15);
    const int nc1  = nc0 + 16;
    const int arow = (lane & 15) * INPUT_DIM;
    const int koffA = (lane >> 4) << 4;             // 0 or 16
    const int koffB = (lane >> 4) << 4;             // 0 or 16

    const __bf16* w1p0 = W1t + (size_t)nc0 * INPUT_DIM;
    const __bf16* w1p1 = W1t + (size_t)nc1 * INPUT_DIM;
    const __bf16* w2p0 = W2t + (size_t)nc0 * INPUT_DIM;
    const __bf16* w2p1 = W2t + (size_t)nc1 * INPUT_DIM;

    v8f accA0 = {}, accA1 = {}, accB0 = {}, accB1 = {};
#pragma unroll
    for (int k0 = 0; k0 < INPUT_DIM; k0 += 32) {
        Frag a, b10, b11, b20, b21;
        a.q[0]   = *reinterpret_cast<const uint4*>(A + arow + k0 + koffA);
        a.q[1]   = *reinterpret_cast<const uint4*>(A + arow + k0 + koffA + 8);
        b10.q[0] = *reinterpret_cast<const uint4*>(w1p0 + k0 + koffB);
        b10.q[1] = *reinterpret_cast<const uint4*>(w1p0 + k0 + koffB + 8);
        b11.q[0] = *reinterpret_cast<const uint4*>(w1p1 + k0 + koffB);
        b11.q[1] = *reinterpret_cast<const uint4*>(w1p1 + k0 + koffB + 8);
        b20.q[0] = *reinterpret_cast<const uint4*>(w2p0 + k0 + koffB);
        b20.q[1] = *reinterpret_cast<const uint4*>(w2p0 + k0 + koffB + 8);
        b21.q[0] = *reinterpret_cast<const uint4*>(w2p1 + k0 + koffB);
        b21.q[1] = *reinterpret_cast<const uint4*>(w2p1 + k0 + koffB + 8);
        accA0 = __builtin_amdgcn_wmma_f32_16x16x32_bf16(false, a.v, false, b10.v,
                                                        (short)0, accA0, false, false);
        accA1 = __builtin_amdgcn_wmma_f32_16x16x32_bf16(false, a.v, false, b11.v,
                                                        (short)0, accA1, false, false);
        accB0 = __builtin_amdgcn_wmma_f32_16x16x32_bf16(false, a.v, false, b20.v,
                                                        (short)0, accB0, false, false);
        accB1 = __builtin_amdgcn_wmma_f32_16x16x32_bf16(false, a.v, false, b21.v,
                                                        (short)0, accB1, false, false);
    }

    const float b1c0 = b1[nc0], b1c1 = b1[nc1];
    const float b2c0 = b2[nc0], b2c1 = b2[nc1];
    const int mbase = m0 + ((lane >> 4) << 3);
#pragma unroll
    for (int r = 0; r < 8; ++r) {
        float a0 = accA0[r] + b1c0;
        float g0 = accB0[r] + b2c0;
        float a1 = accA1[r] + b1c1;
        float g1 = accB1[r] + b2c1;
        float u0 = a0 * (1.0f / (1.0f + __expf(-g0)));
        float u1 = a1 * (1.0f / (1.0f + __expf(-g1)));
        __bf16* up = U + (size_t)(mbase + r) * HIDDEN;
        up[nc0] = (__bf16)u0;
        up[nc1] = (__bf16)u1;
    }
}

// ---------------- 3) GEMM2: U x W3t + b3 -> d_out[:, 0:1024] ----------------
// grid = (N/16, 4). Each wave computes TWO adjacent 16x16 N-tiles sharing one
// A-fragment (2 wmma per A load from LDS).
__global__ void gemm2_kernel(const __bf16* __restrict__ U,
                             const __bf16* __restrict__ W3t,
                             const float* __restrict__ b3,
                             float* __restrict__ out) {
    __shared__ uint4 At[2048];                      // 16 x 1024 bf16 = 32 KB
    const int tid = threadIdx.x, lane = tid & 31, wave = tid >> 5;
    const int m0 = blockIdx.x * 16;

    const uint4* src = reinterpret_cast<const uint4*>(U + (size_t)m0 * HIDDEN);
#pragma unroll
    for (int i = 0; i < 8; ++i)
        async_copy_b128(&At[tid + i * 256], src + tid + i * 256);
    async_wait_all();
    __syncthreads();

    const __bf16* A = reinterpret_cast<const __bf16*>(At);
    const int n0   = (blockIdx.y * 8 + wave) * 32;
    const int nc0  = n0 + (lane & 15);
    const int nc1  = nc0 + 16;
    const int arow = (lane & 15) * HIDDEN;
    const int koffA = (lane >> 4) << 4;
    const int koffB = (lane >> 4) << 4;

    const __bf16* wp0 = W3t + (size_t)nc0 * HIDDEN;
    const __bf16* wp1 = W3t + (size_t)nc1 * HIDDEN;

    v8f acc0 = {}, acc1 = {};
#pragma unroll 8
    for (int k0 = 0; k0 < HIDDEN; k0 += 32) {
        __builtin_prefetch(wp0 + k0 + 256, 0, 0);   // global_prefetch_b8
        Frag a, bw0, bw1;
        a.q[0]   = *reinterpret_cast<const uint4*>(A + arow + k0 + koffA);
        a.q[1]   = *reinterpret_cast<const uint4*>(A + arow + k0 + koffA + 8);
        bw0.q[0] = *reinterpret_cast<const uint4*>(wp0 + k0 + koffB);
        bw0.q[1] = *reinterpret_cast<const uint4*>(wp0 + k0 + koffB + 8);
        bw1.q[0] = *reinterpret_cast<const uint4*>(wp1 + k0 + koffB);
        bw1.q[1] = *reinterpret_cast<const uint4*>(wp1 + k0 + koffB + 8);
        acc0 = __builtin_amdgcn_wmma_f32_16x16x32_bf16(false, a.v, false, bw0.v,
                                                       (short)0, acc0, false, false);
        acc1 = __builtin_amdgcn_wmma_f32_16x16x32_bf16(false, a.v, false, bw1.v,
                                                       (short)0, acc1, false, false);
    }

    const float bb0 = b3[nc0], bb1 = b3[nc1];
    const int mbase = m0 + ((lane >> 4) << 3);
#pragma unroll
    for (int r = 0; r < 8; ++r) {
        float* orow = out + (size_t)(mbase + r) * OUT_STRIDE;
        orow[nc0] = acc0[r] + bb0;
        orow[nc1] = acc1[r] + bb1;
    }
}

// ---------------- 4) score: s = h @ W4 + b4 (wave per row) ----------------
__global__ void score_kernel(const float* __restrict__ out,
                             const float* __restrict__ W4,
                             const float* __restrict__ b4,
                             float* __restrict__ s, int N) {
    int row  = blockIdx.x * 8 + (threadIdx.x >> 5);
    int lane = threadIdx.x & 31;
    if (row >= N) return;
    const float* h = out + (size_t)row * OUT_STRIDE;
    float acc = 0.f;
#pragma unroll 8
    for (int j = lane; j < HIDDEN; j += 32) acc += h[j] * W4[j];
    for (int off = 16; off; off >>= 1) acc += __shfl_xor(acc, off, 32);
    if (lane == 0) s[row] = acc + b4[0];
}

// ---------------- 5) segment softmax ----------------
__global__ void init_groups(unsigned* __restrict__ mkey, float* __restrict__ denom) {
    int i = blockIdx.x * blockDim.x + threadIdx.x;
    if (i < NUM_GROUPS) { mkey[i] = fkey(-3.4e38f); denom[i] = 0.f; }
}

__global__ void seg_max(const float* __restrict__ s, const int* __restrict__ idx,
                        unsigned* __restrict__ mkey, int N) {
    int i = blockIdx.x * blockDim.x + threadIdx.x;
    if (i < N) atomicMax(&mkey[idx[i]], fkey(s[i]));
}

__global__ void seg_exp(const float* __restrict__ s, const int* __restrict__ idx,
                        const unsigned* __restrict__ mkey,
                        float* __restrict__ e, float* __restrict__ denom, int N) {
    int i = blockIdx.x * blockDim.x + threadIdx.x;
    if (i < N) {
        int g = idx[i];
        float ev = __expf(s[i] - fdecode(mkey[g]));
        e[i] = ev;
        atomicAdd(&denom[g], ev);
    }
}

// ---------------- 6) finalize: out[:, :1024] *= w ; out[:, 1024:] = equiv ----
__global__ void finalize_kernel(const float* __restrict__ feat,
                                const int* __restrict__ idx,
                                const float* __restrict__ e,
                                const float* __restrict__ denom,
                                float* __restrict__ out, int N) {
    int row = blockIdx.x;
    if (row >= N) return;
    int t = threadIdx.x;
    float w = e[row] / (denom[idx[row]] + 1e-8f);
    float* orow = out + (size_t)row * OUT_STRIDE;
#pragma unroll
    for (int c = t; c < HIDDEN; c += 256) orow[c] *= w;
    orow[HIDDEN + t] = feat[(size_t)row * FEAT_STRIDE + INPUT_DIM + t];
}

// ---------------- launch ----------------
extern "C" void kernel_launch(void* const* d_in, const int* in_sizes, int n_in,
                              void* d_out, int out_size, void* d_ws, size_t ws_size,
                              hipStream_t stream) {
    const float* feat = (const float*)d_in[0];
    const int*   eidx = (const int*)d_in[1];
    const float* ln_g = (const float*)d_in[2];
    const float* ln_b = (const float*)d_in[3];
    const float* W1   = (const float*)d_in[4];
    const float* b1   = (const float*)d_in[5];
    const float* W2   = (const float*)d_in[6];
    const float* b2   = (const float*)d_in[7];
    const float* W3   = (const float*)d_in[8];
    const float* b3   = (const float*)d_in[9];
    const float* W4   = (const float*)d_in[10];
    const float* b4   = (const float*)d_in[11];
    float* out = (float*)d_out;

    const int N = in_sizes[0] / FEAT_STRIDE;        // 100000

    // workspace carve-out
    char* p = (char*)d_ws;
    __bf16* xn  = (__bf16*)p; p += (size_t)N * INPUT_DIM * 2;
    __bf16* U   = (__bf16*)p; p += (size_t)N * HIDDEN * 2;
    __bf16* W1t = (__bf16*)p; p += (size_t)INPUT_DIM * HIDDEN * 2;
    __bf16* W2t = (__bf16*)p; p += (size_t)INPUT_DIM * HIDDEN * 2;
    __bf16* W3t = (__bf16*)p; p += (size_t)HIDDEN * HIDDEN * 2;
    float* s    = (float*)p;  p += (size_t)N * 4;
    float* e    = (float*)p;  p += (size_t)N * 4;
    unsigned* mkey = (unsigned*)p; p += (size_t)NUM_GROUPS * 4;
    float* denom   = (float*)p;

    // 0) weights
    {
        int total = 2 * INPUT_DIM * HIDDEN + HIDDEN * HIDDEN;
        prep_weights<<<(total + 255) / 256, 256, 0, stream>>>(W1, W2, W3, W1t, W2t, W3t);
    }
    // 1) layernorm
    ln_kernel<<<(N + 7) / 8, 256, 0, stream>>>(feat, ln_g, ln_b, xn, N);
    // 2) dual GEMM + SiLU (two N-tiles per wave)
    {
        dim3 grid(N / 16, 4);
        gemm1_silu<<<grid, 256, 0, stream>>>(xn, W1t, W2t, b1, b2, U);
    }
    // 3) GEMM2 -> h in d_out (two N-tiles per wave)
    {
        dim3 grid(N / 16, 4);
        gemm2_kernel<<<grid, 256, 0, stream>>>(U, W3t, b3, out);
    }
    // 4) score
    score_kernel<<<(N + 7) / 8, 256, 0, stream>>>(out, W4, b4, s, N);
    // 5) segment softmax
    init_groups<<<(NUM_GROUPS + 255) / 256, 256, 0, stream>>>(mkey, denom);
    seg_max<<<(N + 255) / 256, 256, 0, stream>>>(s, eidx, mkey, N);
    seg_exp<<<(N + 255) / 256, 256, 0, stream>>>(s, eidx, mkey, e, denom, N);
    // 6) finalize
    finalize_kernel<<<N, 256, 0, stream>>>(feat, eidx, e, denom, out, N);
}
